// KMeansIterHead_90778428768745
// MI455X (gfx1250) — matrix operand order repaired and checked by reading, not compiled, not run on hardware
//
#include <hip/hip_runtime.h>
#include <hip/hip_bf16.h>

#define DIM     384
#define KCLUST  512
#define THRESHF 500.0f

typedef __attribute__((ext_vector_type(16))) __bf16 v16bf;
typedef __attribute__((ext_vector_type(8)))  float  v8f;

union Frag { v16bf v; unsigned u[8]; };

// ---------------------------------------------------------------------------
// init: zero counts + accumulators, convert centers f32 -> bf16 (L2 resident)
// ---------------------------------------------------------------------------
__global__ void kmh_init_kernel(const float* __restrict__ centers,
                                __hip_bfloat16* __restrict__ cbf,
                                float* __restrict__ accum,
                                float* __restrict__ counts) {
    int tid = blockIdx.x * blockDim.x + threadIdx.x;
    if (tid < KCLUST * DIM) {
        cbf[tid]   = __float2bfloat16(centers[tid]);
        accum[tid] = 0.0f;
    }
    if (tid < KCLUST) counts[tid] = 0.0f;
}

// ---------------------------------------------------------------------------
// assign: WMMA bf16 GEMM (rows x 512 centers, K=384) + per-row argmax.
// Block = 128 threads = 4 waves; 64 rows per block staged in LDS as bf16.
// Argmax is invariant to the per-row L2 norm, so the GEMM uses raw features.
// ---------------------------------------------------------------------------
__launch_bounds__(128)
__global__ void kmh_assign_kernel(const float* __restrict__ feats,
                                  const unsigned* __restrict__ cbf_u32, // bf16 pairs
                                  const int* __restrict__ pseudo,
                                  int* __restrict__ labels,
                                  float* __restrict__ out_labels,
                                  float* __restrict__ out_segs) {
    __shared__ __hip_bfloat16 sA[64 * DIM]; // 48 KB

    const int tid = threadIdx.x;
    const size_t rowBase = (size_t)blockIdx.x * 64;

    // Stage 64x384 f32 -> bf16 LDS, coalesced float4 loads
    const float4* src = (const float4*)(feats + rowBase * DIM);
    for (int i = tid; i < (64 * DIM) / 4; i += 128) {
        float4 v = src[i];
        int b = i * 4;
        sA[b + 0] = __float2bfloat16(v.x);
        sA[b + 1] = __float2bfloat16(v.y);
        sA[b + 2] = __float2bfloat16(v.z);
        sA[b + 3] = __float2bfloat16(v.w);
    }
    __syncthreads();

    const int lane  = tid & 31;
    const int m0    = (tid >> 5) * 16;   // wave's 16-row stripe within block
    const int khalf = lane >> 4;         // ISA 16-bit layout: lane halves split K
    const int nn    = lane & 15;

    float best[8];
    int   bidx[8];
#pragma unroll
    for (int r = 0; r < 8; ++r) { best[r] = -3.402823e38f; bidx[r] = 0; }

    for (int ct = 0; ct < KCLUST / 16; ++ct) {
        const int c0 = ct * 16;
        v8f acc = {0.f, 0.f, 0.f, 0.f, 0.f, 0.f, 0.f, 0.f};
#pragma unroll
        for (int kt = 0; kt < DIM / 32; ++kt) {
            const int kb = kt * 32;
            Frag a, b;
#pragma unroll
            for (int r = 0; r < 8; ++r) {
                // A 16x32 bf16 layout (ISA 7.12.2): lanes0-15 K=0..7,16..23;
                // lanes16-31 K=8..15,24..31 ; 2 bf16 per VGPR.
                int offA = ((r >> 2) << 4) + (khalf << 3) + ((r & 3) << 1);
                a.u[r] = *(const unsigned*)&sA[(m0 + nn) * DIM + kb + offA];
                // B 32x16 bf16: lane = column, same K order as A (lanes0-15 K=0..7,16..23; lanes16-31 K=8..15,24..31)
                int offB = ((r >> 2) << 4) + (khalf << 3) + ((r & 3) << 1);
                b.u[r] = cbf_u32[((c0 + nn) * DIM + kb + offB) >> 1];
            }
            acc = __builtin_amdgcn_wmma_f32_16x16x32_bf16(
                false, a.v, false, b.v, (short)0, acc, false, false);
        }
        // C layout: VGPR r -> row (m0 + khalf*8 + r), lane nn -> column c0+nn
#pragma unroll
        for (int r = 0; r < 8; ++r) {
            float v = acc[r];
            if (v > best[r]) { best[r] = v; bidx[r] = c0 + nn; } // keep-first on ties
        }
    }

    // Reduce argmax across the 16 lanes holding one row (xor<=8 stays in-half)
#pragma unroll
    for (int r = 0; r < 8; ++r) {
        for (int off = 8; off >= 1; off >>= 1) {
            float ov = __shfl_xor(best[r], off, 32);
            int   oi = __shfl_xor(bidx[r], off, 32);
            if (ov > best[r] || (ov == best[r] && oi < bidx[r])) {
                best[r] = ov; bidx[r] = oi;
            }
        }
    }

    if (nn == 0) { // lanes 0 and 16 own rows m0..m0+7 and m0+8..m0+15
        size_t row0 = rowBase + m0 + khalf * 8;
#pragma unroll
        for (int r = 0; r < 8; ++r) {
            size_t rr  = row0 + r;
            int    lab = bidx[r];
            labels[rr]     = lab;
            out_labels[rr] = (float)lab;
            out_segs[rr]   = (float)pseudo[lab];
        }
    }
}

// ---------------------------------------------------------------------------
// accum: one wave per row. Compute ||f||, scatter f/||f|| into accum[label]
// with global f32 atomics (accumulator is 786KB -> L2 resident), bump counts.
// ---------------------------------------------------------------------------
__launch_bounds__(256)
__global__ void kmh_accum_kernel(const float* __restrict__ feats,
                                 const int* __restrict__ labels,
                                 float* __restrict__ counts,
                                 float* __restrict__ accum,
                                 int N) {
    int wave = (blockIdx.x * blockDim.x + threadIdx.x) >> 5;
    int lane = threadIdx.x & 31;
    if (wave >= N) return;

    const float* row = feats + (size_t)wave * DIM;
    float v[DIM / 32];
    float ss = 0.0f;
#pragma unroll
    for (int i = 0; i < DIM / 32; ++i) {
        float x = row[lane + 32 * i];
        v[i] = x;
        ss += x * x;
    }
#pragma unroll
    for (int off = 16; off >= 1; off >>= 1) ss += __shfl_xor(ss, off, 32);

    float scale = 1.0f / fmaxf(sqrtf(ss), 1e-12f);
    int lab = labels[wave];
    if (lane == 0) atomicAdd(counts + lab, 1.0f);
    float* arow = accum + (size_t)lab * DIM;
#pragma unroll
    for (int i = 0; i < DIM / 32; ++i)
        atomicAdd(arow + lane + 32 * i, v[i] * scale);
}

// ---------------------------------------------------------------------------
// finalize: one wave per cluster. normalize(update), blend by (count>500),
// normalize again, write new centers.
// ---------------------------------------------------------------------------
__launch_bounds__(256)
__global__ void kmh_finalize_kernel(const float* __restrict__ accum,
                                    const float* __restrict__ counts,
                                    const float* __restrict__ centers,
                                    float* __restrict__ out) {
    int k    = (blockIdx.x * blockDim.x + threadIdx.x) >> 5;
    int lane = threadIdx.x & 31;
    if (k >= KCLUST) return;

    float v[DIM / 32];
    float ss = 0.0f;
#pragma unroll
    for (int i = 0; i < DIM / 32; ++i) {
        float x = accum[(size_t)k * DIM + lane + 32 * i];
        v[i] = x;
        ss += x * x;
    }
#pragma unroll
    for (int off = 16; off >= 1; off >>= 1) ss += __shfl_xor(ss, off, 32);
    float inv = 1.0f / fmaxf(sqrtf(ss), 1e-12f);

    float uf = (counts[k] > THRESHF) ? 1.0f : 0.0f;
    float ss2 = 0.0f;
#pragma unroll
    for (int i = 0; i < DIM / 32; ++i) {
        float b = v[i] * inv * uf + centers[(size_t)k * DIM + lane + 32 * i] * (1.0f - uf);
        v[i] = b;
        ss2 += b * b;
    }
#pragma unroll
    for (int off = 16; off >= 1; off >>= 1) ss2 += __shfl_xor(ss2, off, 32);
    float inv2 = 1.0f / fmaxf(sqrtf(ss2), 1e-12f);

#pragma unroll
    for (int i = 0; i < DIM / 32; ++i)
        out[(size_t)k * DIM + lane + 32 * i] = v[i] * inv2;
}

// ---------------------------------------------------------------------------
extern "C" void kernel_launch(void* const* d_in, const int* in_sizes, int n_in,
                              void* d_out, int out_size, void* d_ws, size_t ws_size,
                              hipStream_t stream) {
    const float* feats   = (const float*)d_in[0];
    const float* centers = (const float*)d_in[1];
    const int*   pseudo  = (const int*)d_in[2];

    const int N = in_sizes[0] / DIM;   // 393216 rows
    float* out = (float*)d_out;

    // Workspace layout
    char* ws = (char*)d_ws;
    size_t off = 0;
    int*   labels = (int*)(ws + off);             off += (size_t)N * sizeof(int);
    float* counts = (float*)(ws + off);           off += (size_t)KCLUST * sizeof(float);
    float* accum  = (float*)(ws + off);           off += (size_t)KCLUST * DIM * sizeof(float);
    __hip_bfloat16* cbf = (__hip_bfloat16*)(ws + off);

    // 1) zero accumulators (ws is poisoned between runs) + bf16 centers
    kmh_init_kernel<<<(KCLUST * DIM + 255) / 256, 256, 0, stream>>>(
        centers, cbf, accum, counts);

    // 2) WMMA assignment: N/64 blocks of 4 waves
    kmh_assign_kernel<<<N / 64, 128, 0, stream>>>(
        feats, (const unsigned*)cbf, pseudo, labels, out, out + N);

    // 3) normalized scatter-accumulate: one wave per row
    kmh_accum_kernel<<<(N * 32) / 256, 256, 0, stream>>>(
        feats, labels, counts, accum, N);

    // 4) centroid finalize: one wave per cluster
    kmh_finalize_kernel<<<(KCLUST * 32) / 256, 256, 0, stream>>>(
        accum, counts, centers, out + (size_t)2 * N);
}
